// MSAStack_46918222742022
// MI455X (gfx1250) — matrix operand order repaired and checked by reading, not compiled, not run on hardware
//
#include <hip/hip_runtime.h>
#include <cstdint>

// ---------------------------------------------------------------------------
// MSA stack (AlphaFold-style) for MI455X / gfx1250: wave32 + WMMA f16->f32,
// async global->LDS staging (CDNA5 ASYNCcnt path), double-buffered GEMM.
// B=1, S=128, R=256, D=256, Dp=128, H=8, C=32.
// ---------------------------------------------------------------------------

typedef __attribute__((ext_vector_type(16))) _Float16 v16h;
typedef __attribute__((ext_vector_type(8)))  _Float16 v8h;
typedef __attribute__((ext_vector_type(8)))  float    v8f;

#define WMMA_F16(a, b, c) \
  __builtin_amdgcn_wmma_f32_16x16x32_f16(false, (a), false, (b), (short)0, (c), false, false)

__device__ __forceinline__ v16h frag_from(const _Float16* lo, const _Float16* hi) {
  v8h l = *(const v8h*)lo;
  v8h h = *(const v8h*)hi;
  v16h r;
#pragma unroll
  for (int i = 0; i < 8; ++i) { r[i] = l[i]; r[i + 8] = h[i]; }
  return r;
}

// CDNA5 async copy: 16B per lane, global -> LDS, bypasses VGPRs (ASYNCcnt).
__device__ __forceinline__ void async_load_b128(uint32_t lds_byte_off, const _Float16* g) {
  asm volatile("global_load_async_to_lds_b128 %0, %1, off"
               :: "v"(lds_byte_off), "v"(g)
               : "memory");
}
__device__ __forceinline__ void wait_async0() {
  asm volatile("s_wait_asynccnt 0x0" ::: "memory");
}

// ---------------------------------------------------------------------------
// f32 -> f16 conversion (weights)
// ---------------------------------------------------------------------------
__global__ void f32_to_f16_kernel(const float* __restrict__ in, _Float16* __restrict__ out, int n) {
  int i = blockIdx.x * blockDim.x + threadIdx.x;
  if (i < n) out[i] = (_Float16)in[i];
}

// ---------------------------------------------------------------------------
// LayerNorm over last dim D; wave per row; optional transposed output
// (row = s*outR + r  ->  out row = r*outS + s).
// ---------------------------------------------------------------------------
__global__ __launch_bounds__(256) void layernorm_kernel(
    const float* __restrict__ x, const float* __restrict__ gw, const float* __restrict__ bw,
    _Float16* __restrict__ out, int nrows, int D, int outR, int outS, int transpose) {
  int row = blockIdx.x * 8 + (threadIdx.x >> 5);
  int lane = threadIdx.x & 31;
  if (row >= nrows) return;
  const float* xr = x + (size_t)row * D;
  float s = 0.f, s2 = 0.f;
  for (int i = lane; i < D; i += 32) { float v = xr[i]; s += v; s2 += v * v; }
#pragma unroll
  for (int o = 16; o > 0; o >>= 1) { s += __shfl_xor(s, o); s2 += __shfl_xor(s2, o); }
  float mu = s / (float)D;
  float inv = rsqrtf(s2 / (float)D - mu * mu + 1e-5f);
  size_t orow = transpose ? ((size_t)(row % outR) * outS + (row / outR)) : (size_t)row;
  _Float16* op = out + orow * D;
  for (int i = lane; i < D; i += 32) op[i] = (_Float16)((xr[i] - mu) * inv * gw[i] + bw[i]);
}

// ---------------------------------------------------------------------------
// pair bias: b[h][q][k] = sum_d Zln[q,k,d] * bw[h,d]   (Dp=128, H=8)
// ---------------------------------------------------------------------------
__global__ __launch_bounds__(256) void pair_bias_kernel(
    const _Float16* __restrict__ Z, const float* __restrict__ bw, float* __restrict__ out) {
  int qk = blockIdx.x * blockDim.x + threadIdx.x;
  if (qk >= 256 * 256) return;
  const _Float16* z = Z + (size_t)qk * 128;
  float acc[8];
#pragma unroll
  for (int h = 0; h < 8; ++h) acc[h] = 0.f;
  for (int d = 0; d < 128; ++d) {
    float zv = (float)z[d];
#pragma unroll
    for (int h = 0; h < 8; ++h) acc[h] += zv * bw[h * 128 + d];
  }
#pragma unroll
  for (int h = 0; h < 8; ++h) out[(size_t)h * 65536 + qk] = acc[h];
}

// ---------------------------------------------------------------------------
// Generic WMMA GEMM: C[M,N] = act(A[M,K] @ W[N,K]^T + bias) + resid
// Block = 256 thr (8 waves, 2x4), block tile 64x128, wave tile 32x32.
// K staged in 32-chunks via async global->LDS, double-buffered.
// M % 64 == 0, N % 128 == 0, K % 32 == 0 (true for all call sites).
// ACT: 0 none, 1 relu, 2 sigmoid.
// ---------------------------------------------------------------------------
template <int ACT>
__global__ __launch_bounds__(256) void gemm_f16_kernel(
    const _Float16* __restrict__ A, const _Float16* __restrict__ W,
    const float* __restrict__ bias, const float* __restrict__ resid,
    float* __restrict__ outF, _Float16* __restrict__ outH, int M, int N, int K) {
  __shared__ _Float16 As[2][64][40];
  __shared__ _Float16 Ws[2][128][40];
  const int tid = threadIdx.x;
  const int bm = blockIdx.x, bn = blockIdx.y;
  // staging coords: A: 64x32 halfs -> 1 chunk/thread; W: 128x32 -> 2 chunks/thread
  const int sr = tid >> 2;            // 0..63
  const int sc = (tid & 3) << 3;      // 0,8,16,24
  const _Float16* aptr  = A + (size_t)(bm * 64 + sr) * K + sc;
  const _Float16* wptr0 = W + (size_t)(bn * 128 + sr) * K + sc;
  const _Float16* wptr1 = wptr0 + (size_t)64 * K;

  const int wid = tid >> 5, lane = tid & 31;
  const int wm = wid >> 2, wn = wid & 3;       // 2 x 4 waves
  const int n = lane & 15, g = lane >> 4;
  v8f c00 = {}, c01 = {}, c10 = {}, c11 = {};

  // prologue prefetch
  async_load_b128((uint32_t)(uintptr_t)&As[0][sr][sc], aptr);
  async_load_b128((uint32_t)(uintptr_t)&Ws[0][sr][sc], wptr0);
  async_load_b128((uint32_t)(uintptr_t)&Ws[0][64 + sr][sc], wptr1);

  int buf = 0;
  for (int k0 = 0; k0 < K; k0 += 32) {
    wait_async0();
    __syncthreads();
    if (k0 + 32 < K) {
      const int nb = buf ^ 1;
      async_load_b128((uint32_t)(uintptr_t)&As[nb][sr][sc], aptr + k0 + 32);
      async_load_b128((uint32_t)(uintptr_t)&Ws[nb][sr][sc], wptr0 + k0 + 32);
      async_load_b128((uint32_t)(uintptr_t)&Ws[nb][64 + sr][sc], wptr1 + k0 + 32);
    }
    v16h a0 = frag_from(&As[buf][wm * 32 + n][8 * g],      &As[buf][wm * 32 + n][16 + 8 * g]);
    v16h a1 = frag_from(&As[buf][wm * 32 + 16 + n][8 * g], &As[buf][wm * 32 + 16 + n][16 + 8 * g]);
    v16h b0 = frag_from(&Ws[buf][wn * 32 + n][8 * g],      &Ws[buf][wn * 32 + n][16 + 8 * g]);
    v16h b1 = frag_from(&Ws[buf][wn * 32 + 16 + n][8 * g], &Ws[buf][wn * 32 + 16 + n][16 + 8 * g]);
    c00 = WMMA_F16(a0, b0, c00);
    c01 = WMMA_F16(a0, b1, c01);
    c10 = WMMA_F16(a1, b0, c10);
    c11 = WMMA_F16(a1, b1, c11);
    buf ^= 1;
  }

  const int rowb = bm * 64 + wm * 32 + 8 * g;
  const int colb = bn * 128 + wn * 32;
#pragma unroll
  for (int mt = 0; mt < 2; ++mt) {
#pragma unroll
    for (int t = 0; t < 2; ++t) {
      v8f c = mt ? (t ? c11 : c10) : (t ? c01 : c00);
      const int gcol = colb + t * 16 + n;
      const float bv = bias ? bias[gcol] : 0.f;
#pragma unroll
      for (int r = 0; r < 8; ++r) {
        const size_t idx = (size_t)(rowb + mt * 16 + r) * N + gcol;
        float v = c[r] + bv;
        if (ACT == 1) v = fmaxf(v, 0.f);
        if (ACT == 2) v = 1.f / (1.f + __expf(-v));
        if (resid) v += resid[idx];
        if (outF) outF[idx] = v;
        if (outH) outH[idx] = (_Float16)v;
      }
    }
  }
}

// ---------------------------------------------------------------------------
// Fused attention per (outer, head): logits in registers, softmax via 16-lane
// shuffles, P staged per-wave in LDS, V^T staged once per block in LDS.
// QKV rows: [outer*NK + q][768] with q|k|v at col offsets 0|256|512, head*32.
// NK = 256 (row attn, pair bias) or 128 (col attn).
// ---------------------------------------------------------------------------
template <int NK, bool HAS_BIAS>
__global__ __launch_bounds__(128) void attn_kernel(
    const _Float16* __restrict__ QKV, const float* __restrict__ bias,
    _Float16* __restrict__ O) {
  constexpr int NJ = NK / 16;
  __shared__ _Float16 vT[32][NK + 8];
  __shared__ _Float16 Pb[4][16][NK + 8];
  const int h = blockIdx.x & 7;
  const int outer = blockIdx.x >> 3;
  const _Float16* base = QKV + (size_t)outer * NK * 768;
  for (int idx = threadIdx.x; idx < NK * 32; idx += 128) {
    int k = idx >> 5, cc = idx & 31;
    vT[cc][k] = base[(size_t)k * 768 + 512 + h * 32 + cc];
  }
  __syncthreads();
  const int wid = threadIdx.x >> 5, lane = threadIdx.x & 31;
  const int n = lane & 15, g = lane >> 4;
  const float scale = 0.17677669529663689f;  // C^-0.5, C=32
  for (int qc = wid; qc < NJ; qc += 4) {
    const _Float16* qrow = base + (size_t)(qc * 16 + n) * 768 + h * 32;
    v16h aq = frag_from(qrow + 8 * g, qrow + 16 + 8 * g);
    v8f S[NJ];
#pragma unroll
    for (int j = 0; j < NJ; ++j) {
      const _Float16* krow = base + (size_t)(j * 16 + n) * 768 + 256 + h * 32;
      v16h bk = frag_from(krow + 8 * g, krow + 16 + 8 * g);
      v8f z = {};
      S[j] = WMMA_F16(aq, bk, z);
    }
    float mx[8], sm[8];
#pragma unroll
    for (int r = 0; r < 8; ++r) { mx[r] = -1e30f; sm[r] = 0.f; }
#pragma unroll
    for (int j = 0; j < NJ; ++j) {
#pragma unroll
      for (int r = 0; r < 8; ++r) {
        float l = S[j][r] * scale;
        if (HAS_BIAS) {
          const int q = qc * 16 + r + 8 * g;
          l += bias[((size_t)h * NK + q) * NK + j * 16 + n];
        }
        S[j][r] = l;
        mx[r] = fmaxf(mx[r], l);
      }
    }
#pragma unroll
    for (int r = 0; r < 8; ++r) {
#pragma unroll
      for (int o = 1; o < 16; o <<= 1) mx[r] = fmaxf(mx[r], __shfl_xor(mx[r], o));
    }
#pragma unroll
    for (int j = 0; j < NJ; ++j) {
#pragma unroll
      for (int r = 0; r < 8; ++r) {
        float p = __expf(S[j][r] - mx[r]);
        S[j][r] = p;
        sm[r] += p;
      }
    }
#pragma unroll
    for (int r = 0; r < 8; ++r) {
#pragma unroll
      for (int o = 1; o < 16; o <<= 1) sm[r] += __shfl_xor(sm[r], o);
      sm[r] = 1.f / sm[r];
    }
#pragma unroll
    for (int j = 0; j < NJ; ++j) {
#pragma unroll
      for (int r = 0; r < 8; ++r)
        Pb[wid][r + 8 * g][j * 16 + n] = (_Float16)(S[j][r] * sm[r]);
    }
    v8f o0 = {}, o1 = {};
#pragma unroll
    for (int kk = 0; kk < NK; kk += 32) {
      v16h ap = frag_from(&Pb[wid][n][kk + 8 * g], &Pb[wid][n][kk + 16 + 8 * g]);
      v16h b0 = frag_from(&vT[n][kk + 8 * g], &vT[n][kk + 16 + 8 * g]);
      v16h b1 = frag_from(&vT[16 + n][kk + 8 * g], &vT[16 + n][kk + 16 + 8 * g]);
      o0 = WMMA_F16(ap, b0, o0);
      o1 = WMMA_F16(ap, b1, o1);
    }
#pragma unroll
    for (int t = 0; t < 2; ++t) {
      v8f oc = t ? o1 : o0;
#pragma unroll
      for (int r = 0; r < 8; ++r) {
        const int q = qc * 16 + r + 8 * g;
        O[((size_t)outer * NK + q) * 256 + h * 32 + t * 16 + n] = (_Float16)oc[r];
      }
    }
  }
}

// ---------------------------------------------------------------------------
// elementwise gate: a *= b (f16, vectorized x8)
// ---------------------------------------------------------------------------
__global__ __launch_bounds__(256) void eltmul_kernel(
    _Float16* __restrict__ a, const _Float16* __restrict__ b, int n8) {
  int i = blockIdx.x * blockDim.x + threadIdx.x;
  if (i >= n8) return;
  v8h x = ((v8h*)a)[i];
  v8h y = ((const v8h*)b)[i];
#pragma unroll
  for (int k = 0; k < 8; ++k) x[k] = (_Float16)(x[k] * y[k]);
  ((v8h*)a)[i] = x;
}

// ---------------------------------------------------------------------------
// node3[s,r,:] = node2[s,r,:] + colout[r,s,:]   (S=128, R=256, D=256)
// ---------------------------------------------------------------------------
__global__ __launch_bounds__(256) void add_tr_kernel(
    const float* __restrict__ a, const float* __restrict__ bt, float* __restrict__ out) {
  size_t i = (size_t)blockIdx.x * blockDim.x + threadIdx.x;
  if (i >= (size_t)128 * 256 * 256) return;
  int d = (int)(i & 255);
  size_t row = i >> 8;
  int r = (int)(row & 255);
  int s = (int)(row >> 8);
  out[i] = a[i] + bt[(((size_t)r * 128 + s) << 8) + d];
}

// ---------------------------------------------------------------------------
// host launch
// ---------------------------------------------------------------------------
extern "C" void kernel_launch(void* const* d_in, const int* in_sizes, int n_in,
                              void* d_out, int out_size, void* d_ws, size_t ws_size,
                              hipStream_t stream) {
  (void)in_sizes; (void)n_in; (void)out_size; (void)ws_size;
  const int S = 128, R = 256, SR = S * R;

  const float* node = (const float*)d_in[0];
  const float* pair = (const float*)d_in[1];

  char* ws = (char*)d_ws;
  size_t off = 0;
  auto take = [&](size_t bytes) -> void* {
    void* p = ws + off;
    off += (bytes + 255) & ~(size_t)255;
    return p;
  };

  // f16 weights
  _Float16* w_rqkv  = (_Float16*)take((size_t)768 * 256 * 2);
  _Float16* w_rgate = (_Float16*)take((size_t)256 * 256 * 2);
  _Float16* w_ro    = (_Float16*)take((size_t)256 * 256 * 2);
  _Float16* w_cqkv  = (_Float16*)take((size_t)768 * 256 * 2);
  _Float16* w_cgate = (_Float16*)take((size_t)256 * 256 * 2);
  _Float16* w_co    = (_Float16*)take((size_t)256 * 256 * 2);
  _Float16* w_t1    = (_Float16*)take((size_t)1024 * 256 * 2);
  _Float16* w_t2    = (_Float16*)take((size_t)256 * 1024 * 2);
  // activations
  _Float16* Mln   = (_Float16*)take((size_t)SR * 256 * 2);
  _Float16* Zln   = (_Float16*)take((size_t)R * R * 128 * 2);  // aliased as Gate later
  float*    biasb = (float*)take((size_t)8 * R * R * 4);
  char*     big   = (char*)take((size_t)SR * 1024 * 2);        // QKV + O  (or h1)
  _Float16* QKVb  = (_Float16*)big;
  _Float16* Ob    = (_Float16*)(big + (size_t)SR * 768 * 2);
  _Float16* h1    = (_Float16*)big;                            // transition phase alias
  float*    node2 = (float*)take((size_t)SR * 256 * 4);
  float*    colout= (float*)take((size_t)SR * 256 * 4);
  float*    node3 = (float*)take((size_t)SR * 256 * 4);
  _Float16* Gate  = Zln;  // Zln dead after pair_bias

  auto conv = [&](const void* src, _Float16* dst, int n) {
    f32_to_f16_kernel<<<(n + 255) / 256, 256, 0, stream>>>((const float*)src, dst, n);
  };
  conv(d_in[7],  w_rqkv,  768 * 256);
  conv(d_in[9],  w_rgate, 256 * 256);
  conv(d_in[11], w_ro,    256 * 256);
  conv(d_in[15], w_cqkv,  768 * 256);
  conv(d_in[17], w_cgate, 256 * 256);
  conv(d_in[19], w_co,    256 * 256);
  conv(d_in[23], w_t1,    1024 * 256);
  conv(d_in[25], w_t2,    256 * 1024);

  // ---- MSARowAttentionWithPairBias ----
  layernorm_kernel<<<SR / 8, 256, 0, stream>>>(node, (const float*)d_in[2], (const float*)d_in[3],
                                               Mln, SR, 256, 0, 0, 0);
  layernorm_kernel<<<(R * R) / 8, 256, 0, stream>>>(pair, (const float*)d_in[4], (const float*)d_in[5],
                                                    Zln, R * R, 128, 0, 0, 0);
  pair_bias_kernel<<<(R * R) / 256, 256, 0, stream>>>(Zln, (const float*)d_in[6], biasb);

  gemm_f16_kernel<0><<<dim3(SR / 64, 6), 256, 0, stream>>>(
      Mln, w_rqkv, (const float*)d_in[8], nullptr, nullptr, QKVb, SR, 768, 256);
  attn_kernel<256, true><<<S * 8, 128, 0, stream>>>(QKVb, biasb, Ob);
  gemm_f16_kernel<2><<<dim3(SR / 64, 2), 256, 0, stream>>>(
      Mln, w_rgate, (const float*)d_in[10], nullptr, nullptr, Gate, SR, 256, 256);
  eltmul_kernel<<<(SR * 256 / 8 + 255) / 256, 256, 0, stream>>>(Ob, Gate, SR * 256 / 8);
  gemm_f16_kernel<0><<<dim3(SR / 64, 2), 256, 0, stream>>>(
      Ob, w_ro, (const float*)d_in[12], node, node2, nullptr, SR, 256, 256);

  // ---- MSAColumnAttention (LN writes transposed [R,S,D] layout) ----
  layernorm_kernel<<<SR / 8, 256, 0, stream>>>(node2, (const float*)d_in[13], (const float*)d_in[14],
                                               Mln, SR, 256, R, S, 1);
  gemm_f16_kernel<0><<<dim3(SR / 64, 6), 256, 0, stream>>>(
      Mln, w_cqkv, (const float*)d_in[16], nullptr, nullptr, QKVb, SR, 768, 256);
  attn_kernel<128, false><<<R * 8, 128, 0, stream>>>(QKVb, nullptr, Ob);
  gemm_f16_kernel<2><<<dim3(SR / 64, 2), 256, 0, stream>>>(
      Mln, w_cgate, (const float*)d_in[18], nullptr, nullptr, Gate, SR, 256, 256);
  eltmul_kernel<<<(SR * 256 / 8 + 255) / 256, 256, 0, stream>>>(Ob, Gate, SR * 256 / 8);
  gemm_f16_kernel<0><<<dim3(SR / 64, 2), 256, 0, stream>>>(
      Ob, w_co, (const float*)d_in[20], nullptr, colout, nullptr, SR, 256, 256);
  add_tr_kernel<<<(SR * 256) / 256, 256, 0, stream>>>(node2, colout, node3);

  // ---- MSATransition ----
  layernorm_kernel<<<SR / 8, 256, 0, stream>>>(node3, (const float*)d_in[21], (const float*)d_in[22],
                                               Mln, SR, 256, 0, 0, 0);
  gemm_f16_kernel<1><<<dim3(SR / 64, 8), 256, 0, stream>>>(
      Mln, w_t1, (const float*)d_in[24], nullptr, nullptr, h1, SR, 1024, 256);
  gemm_f16_kernel<0><<<dim3(SR / 64, 2), 256, 0, stream>>>(
      h1, w_t2, (const float*)d_in[26], node3, (float*)d_out, nullptr, SR, 256, 1024);
}
